// Decoder_35648228557240
// MI455X (gfx1250) — matrix operand order repaired and checked by reading, not compiled, not run on hardware
//
#include <hip/hip_runtime.h>
#include <hip/hip_bf16.h>

// ---------------- problem constants ----------------
constexpr int Hc  = 512;
constexpr int H2  = 1024;     // 2*H
constexpr int Bc  = 64;
constexpr int Lc  = 45;
constexpr int Sc  = 256;
constexpr int Vc  = 4096;
constexpr int LB  = Lc * Bc;   // 2880
constexpr int SB  = Sc * Bc;   // 16384

typedef __attribute__((ext_vector_type(16))) __bf16 v16bf;
typedef __attribute__((ext_vector_type(8)))  __bf16 v8bf;
typedef __attribute__((ext_vector_type(8)))  float  v8f;
typedef __attribute__((ext_vector_type(4)))  unsigned int v4u;
typedef __attribute__((ext_vector_type(8)))  int v8i;
typedef __attribute__((ext_vector_type(4)))  int v4i;

// ---- Tensor Data Mover availability (this toolchain: 6-arg clang-23 form) ----
#if defined(__has_builtin)
#  if __has_builtin(__builtin_amdgcn_tensor_load_to_lds) && \
      __has_builtin(__builtin_amdgcn_s_wait_tensorcnt)
#    define USE_TDM 1
#  endif
#endif
#ifndef USE_TDM
#  define USE_TDM 0
#endif

#if USE_TDM
// Build a 2D tile descriptor and issue TENSOR_LOAD_TO_LDS.
//   gaddr   : byte address of tile start within tensor
//   lds_off : LDS byte offset of destination
//   dim0    : tensor length along contiguous dim (elements)
//   dim1    : tensor length along row dim (elements)
//   t0, t1  : tile dims (elements)
//   stride0 : elements between consecutive rows
__device__ __forceinline__ void tdm_load_2d(const void* gaddr, unsigned lds_off,
                                            unsigned dim0, unsigned dim1,
                                            unsigned t0, unsigned t1,
                                            unsigned stride0) {
    unsigned long long ga = (unsigned long long)gaddr;
    v4u g0;
    g0[0] = 1u;                                           // count=1, user D#
    g0[1] = lds_off;                                      // lds_addr (bytes)
    g0[2] = (unsigned)(ga & 0xffffffffu);                 // global_addr[31:0]
    g0[3] = (unsigned)((ga >> 32) & 0x01ffffffu)          // global_addr[56:32]
          | (2u << 30);                                   // type = 2 ("image")
    v8i g1;
    g1[0] = (int)(1u << 16);                              // data_size=1 -> 2 bytes
    g1[1] = (int)((dim0 & 0xffffu) << 16);                // tensor_dim0[15:0]
    g1[2] = (int)((dim0 >> 16) | ((dim1 & 0xffffu) << 16)); // dim0 hi | dim1 lo
    g1[3] = (int)((dim1 >> 16) | (t0 << 16));             // dim1 hi | tile_dim0
    g1[4] = (int)(t1 & 0xffffu);                          // tile_dim1 (tile_dim2=0)
    g1[5] = (int)stride0;                                 // tensor_dim0_stride lo32
    g1[6] = 0;                                            // stride hi | dim1_stride lo
    g1[7] = 0;
    v4i z4 = {0, 0, 0, 0};
    v8i z8 = {0, 0, 0, 0, 0, 0, 0, 0};
    __builtin_amdgcn_tensor_load_to_lds(g0, g1, z4, z4, z8, 0);
}
#endif

// ---------------- weight convert + transpose: W[K,N] f32 -> Wt[N,K] bf16 ----
__global__ void k_wt_bf16(const float* __restrict__ W, __bf16* __restrict__ Wt,
                          int K, int N) {
    int idx = blockIdx.x * blockDim.x + threadIdx.x;
    if (idx >= K * N) return;
    int k = idx / N, n = idx - k * N;
    Wt[(size_t)n * K + k] = (__bf16)W[idx];
}

// ---------------- embedding gather: out[b,h] = wdv[b, ids[b], h] ------------
__global__ void k_gather_embed(const int* __restrict__ ids,
                               const float* __restrict__ wdv,
                               float* __restrict__ out) {
    int idx = blockIdx.x * blockDim.x + threadIdx.x;
    if (idx >= Bc * Hc) return;
    int b = idx >> 9, h = idx & (Hc - 1);
    out[idx] = wdv[((size_t)b * Vc + (size_t)ids[b]) * Hc + h];
}

// ---------------- parent gather --------------------------------------------
__global__ void k_parent(const int* __restrict__ pidx,
                         const float* __restrict__ nwdv,
                         const float* __restrict__ lemb,
                         float* __restrict__ out) {
    int idx = blockIdx.x * blockDim.x + threadIdx.x;
    if (idx >= Bc * Hc) return;
    int b = idx >> 9, h = idx & (Hc - 1);
    int l = pidx[b];
    out[idx] = (l < Lc - 1) ? nwdv[((size_t)l * Bc + b) * Hc + h]
                            : lemb[(size_t)b * Hc + h];
}

// ---------------- concat to bf16: dst[M,2H] = [left | right] ----------------
__global__ void k_concat_bf16(__bf16* __restrict__ dst,
                              const float* __restrict__ lft, int lmod,
                              const float* __restrict__ rgt, int rmod, int M) {
    int idx = blockIdx.x * blockDim.x + threadIdx.x;
    if (idx >= M * H2) return;
    int row = idx >> 10, j = idx & (H2 - 1);
    float v = (j < Hc) ? lft[(size_t)(row % lmod) * Hc + j]
                       : rgt[(size_t)(row % rmod) * Hc + (j - Hc)];
    dst[idx] = (__bf16)v;
}

// ---------------- zero fill -------------------------------------------------
__global__ void k_zero(float* __restrict__ p, int n) {
    int idx = blockIdx.x * blockDim.x + threadIdx.x;
    if (idx < n) p[idx] = 0.f;
}

// ---------------- LDS-tiled bf16 WMMA GEMM ----------------------------------
// C[M,N] = act(A[M,K] @ Wt[N,K]^T + bias)
// 64x64 macro-tile, 8 waves; wave w: rows (w&3)*16, cols ((w>>2)*2)*16 + {0,16}.
// K staged in 64-wide chunks in LDS (TDM when available, sync staging else).
// ACT: 0=none 1=relu 2=sigmoid 3=tanh-dot(Ws)->atomic scores[M]
template <int ACT>
__global__ __launch_bounds__(256)
void k_gemm_tiled(const __bf16* __restrict__ A, const __bf16* __restrict__ Wt,
                  const float* __restrict__ bias, float* __restrict__ out,
                  const float* __restrict__ wsv, float* __restrict__ scores,
                  int M, int N, int K) {
    __shared__ __align__(16) __bf16 smem[2 * 64 * 64];   // sA | sB, 8KB each
    __bf16* sA = smem;            // [m 0..63][k 0..63], pitch 64
    __bf16* sB = smem + 64 * 64;  // [n 0..63][k 0..63], pitch 64

    int nBlocks = N >> 6;
    int bM = blockIdx.x / nBlocks, bN = blockIdx.x - bM * nBlocks;
    int mBase = bM << 6, nBase = bN << 6;
    int tid  = threadIdx.x;
    int wave = tid >> 5, lane = tid & 31;
    int half = lane >> 4, ml = lane & 15;
    int mSub = wave & 3, nSub = (wave >> 2) << 1;

    v8f acc0 = {}, acc1 = {};

    for (int k0 = 0; k0 < K; k0 += 64) {
        __syncthreads();   // previous chunk fully consumed before overwrite
#if USE_TDM
        if (wave == 0) {
            tdm_load_2d(A  + (size_t)mBase * K + k0, 0u,
                        (unsigned)K, (unsigned)M, 64u, 64u, (unsigned)K);
            tdm_load_2d(Wt + (size_t)nBase * K + k0, 64u * 64u * 2u,
                        (unsigned)K, (unsigned)N, 64u, 64u, (unsigned)K);
            __builtin_amdgcn_s_wait_tensorcnt(0);
        }
#else
        for (int id = tid; id < 512; id += 256) {          // A: 64 rows x 8 xfers
            int r = id >> 3, c = (id & 7) << 3;
            *(v8bf*)&sA[r * 64 + c] = *(const v8bf*)(A + (size_t)(mBase + r) * K + k0 + c);
        }
        for (int id = tid; id < 512; id += 256) {          // B
            int r = id >> 3, c = (id & 7) << 3;
            *(v8bf*)&sB[r * 64 + c] = *(const v8bf*)(Wt + (size_t)(nBase + r) * K + k0 + c);
        }
#endif
        __syncthreads();

#pragma unroll
        for (int kc = 0; kc < 64; kc += 32) {
            // A fragment (16x32): lane half h holds K = 16h+0..15
            const __bf16* ap = &sA[(mSub * 16 + ml) * 64 + kc + half * 16];
            v8bf alo = *(const v8bf*)ap;
            v8bf ahi = *(const v8bf*)(ap + 8);
            // B fragments (32x16): lane half h holds K = 16h+0..15
            const __bf16* bp0 = &sB[(nSub * 16 + ml) * 64 + kc + half * 16];
            const __bf16* bp1 = bp0 + 16 * 64;
            v8bf b0lo = *(const v8bf*)bp0, b0hi = *(const v8bf*)(bp0 + 8);
            v8bf b1lo = *(const v8bf*)bp1, b1hi = *(const v8bf*)(bp1 + 8);
            v16bf a, b0, b1;
#pragma unroll
            for (int i = 0; i < 8; ++i) {
                a[i] = alo[i];  a[i + 8] = ahi[i];
                b0[i] = b0lo[i]; b0[i + 8] = b0hi[i];
                b1[i] = b1lo[i]; b1[i + 8] = b1hi[i];
            }
            acc0 = __builtin_amdgcn_wmma_f32_16x16x32_bf16(
                false, a, false, b0, (short)0, acc0, false, false);
            acc1 = __builtin_amdgcn_wmma_f32_16x16x32_bf16(
                false, a, false, b1, (short)0, acc1, false, false);
        }
    }

    int col0 = nBase + nSub * 16 + ml;
    int col1 = col0 + 16;
    float bv0 = bias ? bias[col0] : 0.f;
    float bv1 = bias ? bias[col1] : 0.f;

    if (ACT == 3) {
        float w0 = wsv[col0], w1 = wsv[col1];
#pragma unroll
        for (int r = 0; r < 8; ++r) {
            int row = mBase + mSub * 16 + half * 8 + r;
            float v = tanhf(acc0[r] + bv0) * w0 + tanhf(acc1[r] + bv1) * w1;
            for (int off = 1; off < 16; off <<= 1)     // 16 lanes share a row
                v += __shfl_xor(v, off, 32);
            if (ml == 0) atomicAdd(&scores[row], v);
        }
    } else {
#pragma unroll
        for (int r = 0; r < 8; ++r) {
            int row = mBase + mSub * 16 + half * 8 + r;
            float v0 = acc0[r] + bv0, v1 = acc1[r] + bv1;
            if (ACT == 1) { v0 = v0 > 0.f ? v0 : 0.f; v1 = v1 > 0.f ? v1 : 0.f; }
            if (ACT == 2) { v0 = 1.f / (1.f + __expf(-v0)); v1 = 1.f / (1.f + __expf(-v1)); }
            out[(size_t)row * N + col0] = v0;
            out[(size_t)row * N + col1] = v1;
        }
    }
}

// ---------------- gate stage 1: finput = LN((1-sig(t0))*pin)*g+b ------------
__global__ void k_ln_gate0(const float* __restrict__ t0,
                           const float* __restrict__ pin,
                           const float* __restrict__ g,
                           const float* __restrict__ bb,
                           float* __restrict__ fin) {
    int row = blockIdx.x, tid = threadIdx.x;
    const float* xr = t0  + (size_t)row * Hc;
    const float* pr = pin + (size_t)row * Hc;
    float x0 = xr[tid], x1 = xr[tid + 256];
    float u0 = (1.f - 1.f / (1.f + __expf(-x0))) * pr[tid];
    float u1 = (1.f - 1.f / (1.f + __expf(-x1))) * pr[tid + 256];
    __shared__ float s1[256], s2[256];
    s1[tid] = u0 + u1; s2[tid] = u0 * u0 + u1 * u1;
    __syncthreads();
    for (int st = 128; st > 0; st >>= 1) {
        if (tid < st) { s1[tid] += s1[tid + st]; s2[tid] += s2[tid + st]; }
        __syncthreads();
    }
    float mean = s1[0] * (1.f / Hc);
    float var  = s2[0] * (1.f / Hc) - mean * mean;
    float inv  = rsqrtf(var + 1e-5f);
    fin[(size_t)row * Hc + tid]       = (u0 - mean) * inv * g[tid]       + bb[tid];
    fin[(size_t)row * Hc + tid + 256] = (u1 - mean) * inv * g[tid + 256] + bb[tid + 256];
}

// ---------------- gate stage 2: out = finput - LN(c*z) ----------------------
__global__ void k_ln_gate1(const float* __restrict__ c,
                           const float* __restrict__ z,
                           const float* __restrict__ fin,
                           const float* __restrict__ g,
                           const float* __restrict__ bb,
                           float* __restrict__ out) {
    int row = blockIdx.x, tid = threadIdx.x;
    const float* cr = c + (size_t)row * Hc;
    const float* zr = z + (size_t)row * Hc;
    float u0 = cr[tid] * zr[tid];
    float u1 = cr[tid + 256] * zr[tid + 256];
    __shared__ float s1[256], s2[256];
    s1[tid] = u0 + u1; s2[tid] = u0 * u0 + u1 * u1;
    __syncthreads();
    for (int st = 128; st > 0; st >>= 1) {
        if (tid < st) { s1[tid] += s1[tid + st]; s2[tid] += s2[tid + st]; }
        __syncthreads();
    }
    float mean = s1[0] * (1.f / Hc);
    float var  = s2[0] * (1.f / Hc) - mean * mean;
    float inv  = rsqrtf(var + 1e-5f);
    float f0 = (u0 - mean) * inv * g[tid]       + bb[tid];
    float f1 = (u1 - mean) * inv * g[tid + 256] + bb[tid + 256];
    out[(size_t)row * Hc + tid]       = fin[(size_t)row * Hc + tid]       - f0;
    out[(size_t)row * Hc + tid + 256] = fin[(size_t)row * Hc + tid + 256] - f1;
}

// ---------------- softmax over n, scores stored [n,B], out [B,n] ------------
__global__ void k_softmax(const float* __restrict__ scores,
                          const unsigned char* __restrict__ mask,  // [B,n] or null
                          float* __restrict__ out, int n) {
    int b = blockIdx.x, tid = threadIdx.x;
    __shared__ float red[256];
    float mx = -3.0e38f;
    for (int i = tid; i < n; i += 256) {
        float e = scores[(size_t)i * Bc + b];
        if (mask && mask[(size_t)b * n + i]) e = -1e12f;
        mx = fmaxf(mx, e);
    }
    red[tid] = mx; __syncthreads();
    for (int st = 128; st > 0; st >>= 1) {
        if (tid < st) red[tid] = fmaxf(red[tid], red[tid + st]);
        __syncthreads();
    }
    mx = red[0]; __syncthreads();
    float sum = 0.f;
    for (int i = tid; i < n; i += 256) {
        float e = scores[(size_t)i * Bc + b];
        if (mask && mask[(size_t)b * n + i]) e = -1e12f;
        float p = __expf(e - mx);
        out[(size_t)b * n + i] = p;
        sum += p;
    }
    red[tid] = sum; __syncthreads();
    for (int st = 128; st > 0; st >>= 1) {
        if (tid < st) red[tid] += red[tid + st];
        __syncthreads();
    }
    float invs = 1.f / red[0];
    for (int i = tid; i < n; i += 256) out[(size_t)b * n + i] *= invs;
}

// ---------------- weighted sum: out[b,h] = sum_i w[b,i] * X[i,b,h] ----------
__global__ void k_wsum(const float* __restrict__ w,
                       const float* __restrict__ X,
                       float* __restrict__ out, int n) {
    int idx = blockIdx.x * blockDim.x + threadIdx.x;
    if (idx >= Bc * Hc) return;
    int b = idx >> 9, h = idx & (Hc - 1);
    float acc = 0.f;
    for (int i = 0; i < n; ++i)
        acc += w[(size_t)b * n + i] * X[((size_t)i * Bc + b) * Hc + h];
    out[idx] = acc;
}

// ============================================================================
extern "C" void kernel_launch(void* const* d_in, const int* in_sizes, int n_in,
                              void* d_out, int out_size, void* d_ws, size_t ws_size,
                              hipStream_t stream) {
    (void)in_sizes; (void)n_in; (void)out_size; (void)ws_size;
    auto F = [&](int i) { return (const float*)d_in[i]; };
    auto I = [&](int i) { return (const int*)d_in[i]; };

    const int*   input_ids   = I(0);
    const float* nwdv        = F(1);
    const int*   parent_idx  = I(2);
    const float* last_ctx    = F(3);   // [B,1,H] == [B,H]
    const float* feed_hidden = F(4);   // [L,B,H]
    const float* wdv         = F(5);
    const float* enc         = F(6);   // [S,B,H]
    const unsigned char* seq_mask = (const unsigned char*)d_in[7];
    const float* parent_r    = F(8);

    const float* wsrc[13] = { F(9), F(11), F(13), F(19), F(21), F(23),
                              F(29), F(31), F(33), F(39), F(41), F(43), F(46) };

    float* out_ctx = (float*)d_out;                 // [B,H]
    float* out_par = (float*)d_out + Bc * Hc;       // [B,H]

    // -------- workspace layout ------------------------------------------
    char* w = (char*)d_ws;
    size_t off = 0;
    auto take = [&](size_t nbytes) {
        size_t r = off; off += (nbytes + 255) & ~(size_t)255; return r;
    };
    const size_t WT_B  = (size_t)H2 * Hc * sizeof(__bf16);
    const size_t BH_B  = (size_t)Bc * Hc * sizeof(float);
    const size_t FH_B  = (size_t)LB * Hc * sizeof(float);

    __bf16* wt[13];
    for (int i = 0; i < 13; ++i) wt[i] = (__bf16*)(w + take(WT_B));
    float* last_emb = (float*)(w + take(BH_B));
    float* lc       = (float*)(w + take(BH_B));
    float* remain0  = (float*)(w + take(BH_B));
    float* remain   = (float*)(w + take(BH_B));
    float* pfh      = (float*)(w + take(BH_B));
    float* feed_ctx = (float*)(w + take(BH_B));
    float* fh       = (float*)(w + take(FH_B));
    float* scoresF  = (float*)(w + take((size_t)LB * sizeof(float)));
    float* awF      = (float*)(w + take((size_t)Bc * Lc * sizeof(float)));
    float* scoresE  = (float*)(w + take((size_t)SB * sizeof(float)));
    float* awE      = (float*)(w + take((size_t)Bc * Sc * sizeof(float)));
    const size_t GBUF = ((size_t)LB * Hc * sizeof(float) + 255) & ~(size_t)255;
    size_t arena = take((size_t)SB * H2 * sizeof(__bf16));   // 33.5 MB
    __bf16* a_A    = (__bf16*)(w + arena);
    float*  a_t0   = (float*)(w + arena + 1 * GBUF);
    float*  a_fin  = (float*)(w + arena + 2 * GBUF);
    float*  a_c    = (float*)(w + arena + 3 * GBUF);
    float*  a_z    = (float*)(w + arena + 4 * GBUF);
    __bf16* a_Aenc = (__bf16*)(w + arena);   // reuse whole arena after gates

    // -------- launch helpers --------------------------------------------
    auto cdiv = [](int a, int b) { return (a + b - 1) / b; };
    auto concat = [&](__bf16* dst, const float* l, int lm, const float* r, int rm, int M) {
        k_concat_bf16<<<cdiv(M * H2, 256), 256, 0, stream>>>(dst, l, lm, r, rm, M);
    };
    auto gemm = [&](int act, const __bf16* A, const __bf16* Wt, const float* bias,
                    float* outp, const float* wsv, float* sc, int M) {
        int blocks = (M >> 6) * (Hc >> 6);   // 64x64 macro-tiles
        switch (act) {
        case 0: k_gemm_tiled<0><<<blocks, 256, 0, stream>>>(A, Wt, bias, outp, wsv, sc, M, Hc, H2); break;
        case 1: k_gemm_tiled<1><<<blocks, 256, 0, stream>>>(A, Wt, bias, outp, wsv, sc, M, Hc, H2); break;
        case 2: k_gemm_tiled<2><<<blocks, 256, 0, stream>>>(A, Wt, bias, outp, wsv, sc, M, Hc, H2); break;
        default: k_gemm_tiled<3><<<blocks, 256, 0, stream>>>(A, Wt, bias, outp, wsv, sc, M, Hc, H2); break;
        }
    };
    auto run_gate = [&](const float* pin, const float* ctx, int ctx_mod, int M,
                        const __bf16* WiT, const float* bi,
                        const __bf16* WcT, const float* bc,
                        const __bf16* WzT, const float* bz,
                        const float* g0, const float* b0,
                        const float* g1v, const float* b1v, float* outp) {
        concat(a_A, pin, M, ctx, ctx_mod, M);
        gemm(0, a_A, WiT, bi, a_t0, nullptr, nullptr, M);
        k_ln_gate0<<<M, 256, 0, stream>>>(a_t0, pin, g0, b0, a_fin);
        concat(a_A, a_fin, M, ctx, ctx_mod, M);
        gemm(1, a_A, WcT, bc, a_c, nullptr, nullptr, M);
        concat(a_A, a_c, M, ctx, ctx_mod, M);
        gemm(2, a_A, WzT, bz, a_z, nullptr, nullptr, M);
        k_ln_gate1<<<M, 256, 0, stream>>>(a_c, a_z, a_fin, g1v, b1v, outp);
    };

    // -------- 0) convert + transpose weights to bf16 [N,K] --------------
    for (int i = 0; i < 13; ++i)
        k_wt_bf16<<<cdiv(H2 * Hc, 256), 256, 0, stream>>>(wsrc[i], wt[i], H2, Hc);

    // -------- 1) last_embedded gather ------------------------------------
    k_gather_embed<<<cdiv(Bc * Hc, 256), 256, 0, stream>>>(input_ids, wdv, last_emb);

    // -------- 2) gate g1: lc = MinusGate(last_context, last_embedded) ----
    run_gate(last_ctx, last_emb, Bc, Bc,
             wt[0], F(10), wt[1], F(12), wt[2], F(14),
             F(15), F(16), F(17), F(18), lc);

    // -------- 3) gate remain on feed_hidden[-1] --------------------------
    const float* fh_last_in = feed_hidden + (size_t)(Lc - 1) * Bc * Hc;
    run_gate(fh_last_in, last_emb, Bc, Bc,
             wt[3], F(20), wt[4], F(22), wt[5], F(24),
             F(25), F(26), F(27), F(28), remain0);

    // -------- 4) fuse: remain = relu([remain0|lc] @ fuse_W + fuse_b) -----
    concat(a_A, remain0, Bc, lc, Bc, Bc);
    gemm(1, a_A, wt[9], F(40), remain, nullptr, nullptr, Bc);

    // -------- 5) gate gf over all L: fh[L*B,H] ---------------------------
    run_gate(feed_hidden, remain, Bc, LB,
             wt[6], F(30), wt[7], F(32), wt[8], F(34),
             F(35), F(36), F(37), F(38), fh);

    // -------- 6) parent_embedded -> output -------------------------------
    k_parent<<<cdiv(Bc * Hc, 256), 256, 0, stream>>>(parent_idx, nwdv, last_emb, out_par);

    // -------- 7) pfh = relu([parent_r | fh[-1]] @ combine_lr) ------------
    const float* fh_last = fh + (size_t)(Lc - 1) * Bc * Hc;
    concat(a_A, parent_r, Bc, fh_last, Bc, Bc);
    gemm(1, a_A, wt[10], F(42), pfh, nullptr, nullptr, Bc);

    // -------- 8) feed attention over fh (n = L) --------------------------
    concat(a_A, pfh, Bc, fh, LB, LB);                    // left broadcast over L
    k_zero<<<cdiv(LB, 256), 256, 0, stream>>>(scoresF, LB);
    gemm(3, a_A, wt[12], F(47), nullptr, F(48), scoresF, LB);
    k_softmax<<<Bc, 256, 0, stream>>>(scoresF, nullptr, awF, Lc);
    k_wsum<<<cdiv(Bc * Hc, 256), 256, 0, stream>>>(awF, fh, feed_ctx, Lc);

    // -------- 9) encoder attention (n = S, masked) -----------------------
    concat(a_Aenc, feed_ctx, Bc, enc, SB, SB);           // left broadcast over S
    k_zero<<<cdiv(SB, 256), 256, 0, stream>>>(scoresE, SB);
    gemm(3, a_Aenc, wt[11], F(44), nullptr, F(45), scoresE, SB);
    k_softmax<<<Bc, 256, 0, stream>>>(scoresE, seq_mask, awE, Sc);
    k_wsum<<<cdiv(Bc * Hc, 256), 256, 0, stream>>>(awE, enc, out_ctx, Sc);
}
